// Attention_29583734734990
// MI455X (gfx1250) — matrix operand order repaired and checked by reading, not compiled, not run on hardware
//
#include <hip/hip_runtime.h>
#include <hip/hip_bf16.h>

#define N_HEADS 16
#define D_MODEL 1024
#define D_HEAD  64
#define BATCH   4
#define SEQ     2048

typedef __attribute__((ext_vector_type(16))) __bf16 v16bf;
typedef __attribute__((ext_vector_type(8)))  float  v8f;

union Frag {
    v16bf          v;
    __bf16         b[16];
    unsigned int   u[8];
    unsigned short s[16];
};

// A-matrix (16x32 bf16): lanes 0-15 hold K=0..15, lanes 16-31 hold K=16..31.
__device__ __forceinline__ int kA(int p, int h) {
    return 2 * p + 16 * h;
}
// B-matrix (32x16 bf16): lanes 0-15 hold K=0..15, lanes 16-31 hold K=16..31.
__device__ __forceinline__ int kB(int p, int h) {
    return 2 * p + 16 * h;
}

__device__ __forceinline__ unsigned int pack_bf16(float lo, float hi) {
    union { __bf16 b[2]; unsigned int u; } c;
    c.b[0] = (__bf16)lo;
    c.b[1] = (__bf16)hi;
    return c.u;
}

// CDNA5 async copy: each lane copies 16B global -> LDS, tracked by ASYNCcnt.
__device__ __forceinline__ void async_b128(const void* g, void* l) {
    const unsigned loff = (unsigned)(unsigned long long)l;   // LDS aperture: addr[31:0]
    asm volatile("global_load_async_to_lds_b128 %0, %1, off"
                 :: "v"(loff), "v"(g) : "memory");
}
__device__ __forceinline__ void wait_async0() {
    asm volatile("s_wait_asynccnt 0x0" ::: "memory");
}

// ---------------------------------------------------------------------------
// Prep: transpose + fp32->bf16 convert.  in: [nmat][R][C] f32 -> out: [nmat][C][R] bf16
// ---------------------------------------------------------------------------
__global__ __launch_bounds__(256) void transpose_conv_kernel(
    const float* __restrict__ in, __hip_bfloat16* __restrict__ out,
    int nmat, int R, int C)
{
    const size_t total = (size_t)nmat * R * C;
    for (size_t i = (size_t)blockIdx.x * blockDim.x + threadIdx.x; i < total;
         i += (size_t)gridDim.x * blockDim.x) {
        const size_t mrc = (size_t)R * C;
        const int    m   = (int)(i / mrc);
        const size_t rem = i % mrc;
        const int    r   = (int)(rem / C);
        const int    c   = (int)(rem % C);
        out[(size_t)m * mrc + (size_t)c * R + r] = __float2bfloat16(in[i]);
    }
}

// ---------------------------------------------------------------------------
// Kernel 1: QKV projection.  One wave: 16x64 tile of Q, K, or V for one head.
// Weights pre-transposed to [h][d][m] bf16 so B-fragments load as dwords.
// Q,K stored [b,h,s,d] bf16; V stored transposed [b,h,d,s] bf16.
// ---------------------------------------------------------------------------
__global__ __launch_bounds__(256) void qkv_kernel(
    const float* __restrict__ resid,
    const __hip_bfloat16* __restrict__ Wqt,
    const __hip_bfloat16* __restrict__ Wkt,
    const __hip_bfloat16* __restrict__ Wvt,
    __hip_bfloat16* __restrict__ Qb,
    __hip_bfloat16* __restrict__ Kb,
    __hip_bfloat16* __restrict__ Vt)
{
    const int lane = threadIdx.x & 31;
    const int wave = blockIdx.x * (blockDim.x >> 5) + (threadIdx.x >> 5);
    const int RT   = (BATCH * SEQ) / 16;          // 512 row tiles

    const int rt    = wave % RT;
    const int h     = (wave / RT) % N_HEADS;
    const int which = wave / (RT * N_HEADS);      // 0=Q 1=K 2=V
    const int b     = rt / (SEQ / 16);
    const int srow  = (rt % (SEQ / 16)) * 16;

    const int n  = lane & 15;
    const int hi = lane >> 4;

    const __hip_bfloat16* Wt =
        ((which == 0) ? Wqt : (which == 1) ? Wkt : Wvt) + (size_t)h * D_HEAD * D_MODEL;
    const float* arow = resid + ((size_t)b * SEQ + srow + (lane & 15)) * D_MODEL;

    v8f acc[4] = {v8f{}, v8f{}, v8f{}, v8f{}};

    for (int kk = 0; kk < D_MODEL; kk += 32) {
        if (kk + 32 < D_MODEL)
            __builtin_prefetch(arow + kk + 32, 0, 0);

        Frag a;
#pragma unroll
        for (int p = 0; p < 8; ++p) {
            const int kb = kA(p, hi);
            const float2 f = *(const float2*)(arow + kk + kb);
            a.b[2 * p]     = (__bf16)f.x;
            a.b[2 * p + 1] = (__bf16)f.y;
        }
#pragma unroll
        for (int nt = 0; nt < 4; ++nt) {
            Frag bf;
            const unsigned short* wcol =
                (const unsigned short*)(Wt + (size_t)(nt * 16 + n) * D_MODEL);
#pragma unroll
            for (int p = 0; p < 8; ++p)
                bf.u[p] = *(const unsigned int*)(wcol + kk + kB(p, hi));
            acc[nt] = __builtin_amdgcn_wmma_f32_16x16x32_bf16(
                false, a.v, false, bf.v, (short)0, acc[nt], false, false);
        }
    }

    if (which < 2) {
        __hip_bfloat16* dst =
            ((which == 0) ? Qb : Kb) + ((size_t)(b * N_HEADS + h) * SEQ + srow) * D_HEAD;
#pragma unroll
        for (int nt = 0; nt < 4; ++nt) {
#pragma unroll
            for (int r = 0; r < 8; ++r) {
                const int row = hi * 8 + r;
                dst[(size_t)row * D_HEAD + nt * 16 + n] = __float2bfloat16(acc[nt][r]);
            }
        }
    } else {
        __hip_bfloat16* dst = Vt + (size_t)(b * N_HEADS + h) * D_HEAD * SEQ;
#pragma unroll
        for (int nt = 0; nt < 4; ++nt) {
            unsigned short* col =
                (unsigned short*)(dst + (size_t)(nt * 16 + n) * SEQ + srow + hi * 8);
#pragma unroll
            for (int i = 0; i < 4; ++i)
                *(unsigned int*)(col + 2 * i) = pack_bf16(acc[nt][2 * i], acc[nt][2 * i + 1]);
        }
    }
}

// ---------------------------------------------------------------------------
// Kernel 2: causal flash attention.  One block (8 waves) owns a 128-query
// slice of one (b,h): wave w handles query rows qbase+16w..+15.  K (32x64)
// and Vt (64x32) tiles are double-buffered in LDS via async-to-LDS DMA and
// shared by all 8 waves.  Online softmax; P transposed through private LDS.
// ---------------------------------------------------------------------------
__global__ __launch_bounds__(256) void attn_kernel(
    const __hip_bfloat16* __restrict__ Qb,
    const __hip_bfloat16* __restrict__ Kb,
    const __hip_bfloat16* __restrict__ Vt,
    __hip_bfloat16* __restrict__ Z)
{
    __shared__ __align__(16) unsigned short Kl[2][32 * 64];  // 2 x 4KB
    __shared__ __align__(16) unsigned short Vl[2][64 * 32];  // 2 x 4KB
    __shared__ __align__(16) unsigned short Pl[8][16 * 32];  // 8 x 1KB

    const int tid   = threadIdx.x;
    const int lane  = tid & 31;
    const int wslot = tid >> 5;

    const int blk  = blockIdx.x;
    const int qblk = blk & 15;                 // 16 query blocks of 128 rows
    const int h    = (blk >> 4) & 15;
    const int b    = blk >> 8;

    const int qbase = qblk * 128;
    const int qrow0 = qbase + wslot * 16;      // this wave's query tile
    const int nk_wave = qrow0 + 16;            // keys visible to this wave
    const int nk_blk  = qbase + 128;           // loop bound (block max)

    const int n  = lane & 15;
    const int hi = lane >> 4;

    const size_t bh = (size_t)(b * N_HEADS + h);
    const __hip_bfloat16* qbase_p = Qb + (bh * SEQ + (size_t)qrow0) * D_HEAD;
    const __hip_bfloat16* kbase_p = Kb + bh * SEQ * D_HEAD;
    const __hip_bfloat16* vbase_p = Vt + bh * (size_t)D_HEAD * SEQ;

    // K-tile async slice for this thread: row = tid/8 (0..31), seg = tid%8
    const int krow = tid >> 3, kseg = tid & 7;
    // V-tile async slice: d = tid/4 (0..63), seg = tid%4
    const int vd = tid >> 2, vseg = tid & 3;

    // ---- Q fragments (16x64 -> two 16x32 A-fragments) ----
    Frag aq0, aq1;
    {
        const unsigned short* qrow =
            (const unsigned short*)(qbase_p + (size_t)(lane & 15) * D_HEAD);
#pragma unroll
        for (int p = 0; p < 8; ++p) {
            const int kb = kA(p, hi);
            aq0.u[p] = *(const unsigned int*)(qrow + kb);
            aq1.u[p] = *(const unsigned int*)(qrow + 32 + kb);
        }
    }

    v8f o[4] = {v8f{}, v8f{}, v8f{}, v8f{}};
    float mrun[8], lrun[8];
#pragma unroll
    for (int r = 0; r < 8; ++r) { mrun[r] = -1.0e30f; lrun[r] = 0.0f; }

    // ---- prologue: DMA first K/V tile into buffer 0 ----
    async_b128(kbase_p + ((size_t)krow * D_HEAD + kseg * 8), &Kl[0][krow * 64 + kseg * 8]);
    async_b128(vbase_p + ((size_t)vd * SEQ + vseg * 8),      &Vl[0][vd * 32 + vseg * 8]);

    for (int kk = 0; kk < nk_blk; kk += 32) {
        const int cur = (kk >> 5) & 1;

        wait_async0();          // own slice landed in LDS
        __syncthreads();        // all slices visible; prev readers done

        if (kk + 32 < nk_blk) { // DMA next tile into the other buffer
            const int nxt = cur ^ 1;
            async_b128(kbase_p + ((size_t)(kk + 32 + krow) * D_HEAD + kseg * 8),
                       &Kl[nxt][krow * 64 + kseg * 8]);
            async_b128(vbase_p + ((size_t)vd * SEQ + kk + 32 + vseg * 8),
                       &Vl[nxt][vd * 32 + vseg * 8]);
        }

        if (kk < nk_wave) {     // wave-uniform: skip fully-masked steps
            const unsigned short* Kt = Kl[cur];
            const unsigned short* Vtile = Vl[cur];

            // ---- scores: two 16x16 tiles (keys kk..+15, kk+16..+31) ----
            v8f s0 = {}, s1 = {};
#pragma unroll
            for (int j = 0; j < 2; ++j) {
                Frag bk0, bk1;
                const unsigned short* krp = Kt + (j * 16 + n) * 64;
#pragma unroll
                for (int p = 0; p < 8; ++p) {
                    const int kb = kB(p, hi);
                    bk0.u[p] = *(const unsigned int*)(krp + kb);
                    bk1.u[p] = *(const unsigned int*)(krp + 32 + kb);
                }
                v8f& s = j ? s1 : s0;
                s = __builtin_amdgcn_wmma_f32_16x16x32_bf16(false, aq0.v, false, bk0.v, (short)0, s, false, false);
                s = __builtin_amdgcn_wmma_f32_16x16x32_bf16(false, aq1.v, false, bk1.v, (short)0, s, false, false);
            }

            // ---- scale + causal mask ----
#pragma unroll
            for (int r = 0; r < 8; ++r) {
                const int qg = qrow0 + hi * 8 + r;
                float x0 = s0[r] * 0.125f;          // 1/sqrt(64)
                float x1 = s1[r] * 0.125f;
                if (kk + n > qg)      x0 = -1.0e30f;
                if (kk + 16 + n > qg) x1 = -1.0e30f;
                s0[r] = x0;
                s1[r] = x1;
            }

            // ---- online softmax (row reductions over 16-lane half-group) ----
#pragma unroll
            for (int r = 0; r < 8; ++r) {
                float mx = fmaxf(s0[r], s1[r]);
                mx = fmaxf(mx, __shfl_xor(mx, 1, 32));
                mx = fmaxf(mx, __shfl_xor(mx, 2, 32));
                mx = fmaxf(mx, __shfl_xor(mx, 4, 32));
                mx = fmaxf(mx, __shfl_xor(mx, 8, 32));
                const float mnew  = fmaxf(mrun[r], mx);
                const float alpha = __expf(mrun[r] - mnew);
                const float p0 = __expf(s0[r] - mnew);
                const float p1 = __expf(s1[r] - mnew);
                float ps = p0 + p1;
                ps += __shfl_xor(ps, 1, 32);
                ps += __shfl_xor(ps, 2, 32);
                ps += __shfl_xor(ps, 4, 32);
                ps += __shfl_xor(ps, 8, 32);
                lrun[r] = lrun[r] * alpha + ps;
                mrun[r] = mnew;
                s0[r] = p0;
                s1[r] = p1;
                o[0][r] *= alpha;
                o[1][r] *= alpha;
                o[2][r] *= alpha;
                o[3][r] *= alpha;
            }

            // ---- transpose P (C layout -> A fragment) via wave-private LDS ----
            unsigned short* pl = Pl[wslot];
#pragma unroll
            for (int r = 0; r < 8; ++r) {
                const int row = hi * 8 + r;
                union { __bf16 b; unsigned short s; } c0, c1;
                c0.b = (__bf16)s0[r];
                c1.b = (__bf16)s1[r];
                pl[row * 32 + n]      = c0.s;
                pl[row * 32 + 16 + n] = c1.s;
            }
            asm volatile("" ::: "memory");   // DS ops are in-order within a wave
            Frag ap;
            {
                const unsigned short* prow = pl + (lane & 15) * 32;
#pragma unroll
                for (int p = 0; p < 8; ++p)
                    ap.u[p] = *(const unsigned int*)(prow + kA(p, hi));
            }
            asm volatile("" ::: "memory");

            // ---- O += P(16x32) x V(32x64) from LDS Vt tile [d][s] ----
#pragma unroll
            for (int dt = 0; dt < 4; ++dt) {
                Frag bv;
                const unsigned short* vrp = Vtile + (dt * 16 + n) * 32;
#pragma unroll
                for (int p = 0; p < 8; ++p)
                    bv.u[p] = *(const unsigned int*)(vrp + kB(p, hi));
                o[dt] = __builtin_amdgcn_wmma_f32_16x16x32_bf16(
                    false, ap.v, false, bv.v, (short)0, o[dt], false, false);
            }
        }
    }

    // ---- normalize and write Z in [b*s, h*64+d] bf16 ----
    float linv[8];
#pragma unroll
    for (int r = 0; r < 8; ++r) linv[r] = 1.0f / lrun[r];

    __hip_bfloat16* zrow =
        Z + ((size_t)b * SEQ + (size_t)qrow0) * (N_HEADS * D_HEAD) + h * D_HEAD;
#pragma unroll
    for (int dt = 0; dt < 4; ++dt) {
#pragma unroll
        for (int r = 0; r < 8; ++r) {
            zrow[(size_t)(hi * 8 + r) * (N_HEADS * D_HEAD) + dt * 16 + n] =
                __float2bfloat16(o[dt][r] * linv[r]);
        }
    }
}

// ---------------------------------------------------------------------------
// Kernel 3: output projection.  out = Z(8192x1024) x Wo(1024x1024) fp32.
// Wo pre-transposed to [m][k] bf16 so B-fragments load as dwords.
// ---------------------------------------------------------------------------
__global__ __launch_bounds__(256) void out_proj_kernel(
    const __hip_bfloat16* __restrict__ Z,
    const __hip_bfloat16* __restrict__ Wot,
    float* __restrict__ out)
{
    const int lane = threadIdx.x & 31;
    const int wave = blockIdx.x * (blockDim.x >> 5) + (threadIdx.x >> 5);

    const int rt = wave % ((BATCH * SEQ) / 16);   // 0..511
    const int ct = wave / ((BATCH * SEQ) / 16);   // 0..15 (64-col tiles)

    const int n  = lane & 15;
    const int hi = lane >> 4;

    const unsigned short* arow =
        (const unsigned short*)Z + ((size_t)rt * 16 + (lane & 15)) * D_MODEL;

    v8f acc[4] = {v8f{}, v8f{}, v8f{}, v8f{}};

    for (int kk = 0; kk < D_MODEL; kk += 32) {
        if (kk + 32 < D_MODEL)
            __builtin_prefetch(arow + kk + 32, 0, 0);

        Frag a;
#pragma unroll
        for (int p = 0; p < 8; ++p)
            a.u[p] = *(const unsigned int*)(arow + kk + kA(p, hi));
#pragma unroll
        for (int nt = 0; nt < 4; ++nt) {
            Frag bf;
            const unsigned short* wcol =
                (const unsigned short*)(Wot + (size_t)(ct * 64 + nt * 16 + n) * D_MODEL);
#pragma unroll
            for (int p = 0; p < 8; ++p)
                bf.u[p] = *(const unsigned int*)(wcol + kk + kB(p, hi));
            acc[nt] = __builtin_amdgcn_wmma_f32_16x16x32_bf16(
                false, a.v, false, bf.v, (short)0, acc[nt], false, false);
        }
    }

    float* orow = out + (size_t)rt * 16 * D_MODEL + ct * 64;
#pragma unroll
    for (int nt = 0; nt < 4; ++nt) {
#pragma unroll
        for (int r = 0; r < 8; ++r)
            orow[(size_t)(hi * 8 + r) * D_MODEL + nt * 16 + n] = acc[nt][r];
    }
}

// ---------------------------------------------------------------------------
extern "C" void kernel_launch(void* const* d_in, const int* in_sizes, int n_in,
                              void* d_out, int out_size, void* d_ws, size_t ws_size,
                              hipStream_t stream)
{
    (void)in_sizes; (void)n_in; (void)out_size; (void)ws_size;

    const float* resid = (const float*)d_in[0];
    const float* w_q   = (const float*)d_in[1];
    const float* w_k   = (const float*)d_in[2];
    const float* w_v   = (const float*)d_in[3];
    const float* w_o   = (const float*)d_in[4];
    float*       out   = (float*)d_out;

    char* ws = (char*)d_ws;
    const size_t qkv_bytes = (size_t)BATCH * N_HEADS * SEQ * D_HEAD * sizeof(__hip_bfloat16); // 16MB
    const size_t w_bytes   = (size_t)N_HEADS * D_MODEL * D_HEAD * sizeof(__hip_bfloat16);     // 2MB

    __hip_bfloat16* Qb  = (__hip_bfloat16*)(ws);
    __hip_bfloat16* Kb  = (__hip_bfloat16*)(ws + qkv_bytes);
    __hip_bfloat16* Vt  = (__hip_bfloat16*)(ws + 2 * qkv_bytes);
    __hip_bfloat16* Zb  = (__hip_bfloat16*)(ws + 3 * qkv_bytes);
    __hip_bfloat16* Wqt = (__hip_bfloat16*)(ws + 4 * qkv_bytes);
    __hip_bfloat16* Wkt = (__hip_bfloat16*)(ws + 4 * qkv_bytes + 1 * w_bytes);
    __hip_bfloat16* Wvt = (__hip_bfloat16*)(ws + 4 * qkv_bytes + 2 * w_bytes);
    __hip_bfloat16* Wot = (__hip_bfloat16*)(ws + 4 * qkv_bytes + 3 * w_bytes);

    // weight prep: transpose + convert to bf16
    transpose_conv_kernel<<<256, 256, 0, stream>>>(w_q, Wqt, N_HEADS, D_MODEL, D_HEAD);
    transpose_conv_kernel<<<256, 256, 0, stream>>>(w_k, Wkt, N_HEADS, D_MODEL, D_HEAD);
    transpose_conv_kernel<<<256, 256, 0, stream>>>(w_v, Wvt, N_HEADS, D_MODEL, D_HEAD);
    transpose_conv_kernel<<<256, 256, 0, stream>>>(w_o, Wot, 1, D_MODEL, D_MODEL);

    // 512 row-tiles * 16 heads * 3 matrices = 24576 waves -> 3072 blocks
    qkv_kernel<<<3072, 256, 0, stream>>>(resid, Wqt, Wkt, Wvt, Qb, Kb, Vt);
    // 4 batch * 16 heads * 16 query-blocks = 1024 blocks of 8 waves
    attn_kernel<<<1024, 256, 0, stream>>>(Qb, Kb, Vt, Zb);
    // 512*16 = 8192 waves -> 1024 blocks
    out_proj_kernel<<<1024, 256, 0, stream>>>(Zb, Wot, out);
}
